// Decoder_44401371906320
// MI455X (gfx1250) — matrix operand order repaired and checked by reading, not compiled, not run on hardware
//
#include <hip/hip_runtime.h>
#include <hip/hip_bf16.h>
#include <math.h>

// Problem constants (V, M, E, D, B, S) = (50000, 512, 512, 512, 128, 512)
#define VV 50000
#define MM 512
#define EE 512
#define DD 512
#define BB 128
#define SS 512
#define E2 (2 * EE)          // 1024
#define FATT (DD + E2)       // 1536  (attn_W inner dim)
#define FRNN (E2 + MM)       // 1536  (gru_Wih inner dim)
#define G3 (3 * DD)          // 1536
#define FFC (E2 + DD + MM)   // 2048  (fc_W inner dim)

#define AS1 __attribute__((address_space(1)))
#define AS3 __attribute__((address_space(3)))

#if __has_builtin(__builtin_amdgcn_global_load_async_to_lds_b128)
#define HAVE_ASYNC_LDS 1
#else
#define HAVE_ASYNC_LDS 0
#endif

typedef __attribute__((ext_vector_type(16))) __bf16 v16bf;
typedef __attribute__((ext_vector_type(8)))  __bf16 v8bf;
typedef __attribute__((ext_vector_type(4)))  __bf16 v4bf;
typedef __attribute__((ext_vector_type(2)))  __bf16 v2bf;
typedef __attribute__((ext_vector_type(8)))  float  v8f;
typedef __attribute__((ext_vector_type(4)))  float  v4f;
typedef __attribute__((ext_vector_type(2)))  float  v2f;
typedef __attribute__((ext_vector_type(4)))  int    v4i;

__device__ __forceinline__ v4bf cvt4(v4f x) {
    return __builtin_convertvector(x, v4bf);   // packed v_cvt bf16 pairs
}
__device__ __forceinline__ v2bf cvt2(v2f x) {
    return __builtin_convertvector(x, v2bf);
}

__device__ __forceinline__ v8f wmma_bf16(v16bf a, v16bf b, v8f c) {
    return __builtin_amdgcn_wmma_f32_16x16x32_bf16(
        /*neg_a=*/false, a, /*neg_b=*/false, b,
        /*c_mod=*/(short)0, c, /*reuse_a=*/false, /*reuse_b=*/false);
}

// A-fragment: lane holds row m=lane&15; K elems {kb..kb+15},
// kb=(lane>>4)*16 -> two contiguous 8x bf16 (16 B) loads.
__device__ __forceinline__ v16bf load_a_frag(const __bf16* p) {
    v8bf lo = *(const v8bf*)(p);
    v8bf hi = *(const v8bf*)(p + 8);
    return __builtin_shufflevector(lo, hi, 0, 1, 2, 3, 4, 5, 6, 7,
                                   8, 9, 10, 11, 12, 13, 14, 15);
}

// B-fragment from fp32 LDS: 16 contiguous floats (64 B) + packed cvt.
__device__ __forceinline__ v16bf load_b_frag_f32(const float* p) {
    v8f lo = *(const v8f*)(p);
    v8f hi = *(const v8f*)(p + 8);
    v8bf l = __builtin_convertvector(lo, v8bf);
    v8bf h = __builtin_convertvector(hi, v8bf);
    return __builtin_shufflevector(l, h, 0, 1, 2, 3, 4, 5, 6, 7,
                                   8, 9, 10, 11, 12, 13, 14, 15);
}

// ---------------------------------------------------------------------------
// generic fp32 -> bf16 converter (packed), over n/4 quads
// ---------------------------------------------------------------------------
__global__ void cvt_kernel(const float* __restrict__ src,
                           __bf16* __restrict__ dst, int n4) {
    int i = blockIdx.x * 256 + threadIdx.x;
    if (i < n4) *(v4bf*)(dst + 4 * (size_t)i) = cvt4(*(const v4f*)(src + 4 * (size_t)i));
}

// ---------------------------------------------------------------------------
// K0: embedded = emb_table[input] -> bf16 into rnn_in[:,0:512], zcat[:,1536:2048]
// ---------------------------------------------------------------------------
__global__ void embed_kernel(const int* __restrict__ input,
                             const float* __restrict__ emb,
                             __bf16* __restrict__ rnn_bf,
                             __bf16* __restrict__ zcat_bf) {
    int b = blockIdx.y;
    int j = (blockIdx.x * 256 + threadIdx.x) * 2;          // 0..510
    v2bf y = cvt2(*(const v2f*)(emb + (size_t)input[b] * MM + j));
    *(v2bf*)(rnn_bf + (size_t)b * FRNN + j) = y;
    *(v2bf*)(zcat_bf + (size_t)b * FFC + (DD + E2) + j) = y;
}

// ---------------------------------------------------------------------------
// Generic wave GEMM (bf16 in, f32 out): C[m,n] = sum_k A[m,k]*W[n,k] + bias[n]
// grid (N/16, M/16), block 32. A-frag: 2x16B loads; B-frag: one 32B load.
// ---------------------------------------------------------------------------
__global__ void gemm_bf16(const __bf16* __restrict__ A, int lda,
                          const __bf16* __restrict__ W, int ldw,
                          const float* __restrict__ bias,
                          float* __restrict__ C, int ldc, int K) {
    int lane  = threadIdx.x;
    int mtile = blockIdx.y, ntile = blockIdx.x;
    int l15   = lane & 15;
    int kbA   = (lane >> 4) * 16;
    int kbB   = (lane >> 4) * 16;
    const __bf16* arow = A + (size_t)(mtile * 16 + l15) * lda;
    const __bf16* wrow = W + (size_t)(ntile * 16 + l15) * ldw;
    v8f acc = {};
#pragma unroll 4
    for (int k0 = 0; k0 < K; k0 += 32) {
        v16bf af = load_a_frag(arow + k0 + kbA);
        v16bf bf = *(const v16bf*)(wrow + k0 + kbB);
        acc = wmma_bf16(af, bf, acc);
    }
    int rbase = (lane >> 4) * 8;
    float bv  = bias ? bias[ntile * 16 + l15] : 0.0f;
#pragma unroll
    for (int r = 0; r < 8; ++r) {
        int row = mtile * 16 + rbase + r;
        C[(size_t)row * ldc + ntile * 16 + l15] = acc[r] + bv;
    }
}

// ---------------------------------------------------------------------------
// K2: fused attention scores.
// scores[b,s] = sum_d v_w[d] * tanh( enc[b,s,:] . attn_W[d,512:] + hid_proj[b,d] )
// grid (S/16, B), block 128 = 4 waves; enc tile staged as bf16 in LDS.
// ---------------------------------------------------------------------------
__global__ void attn_scores_kernel(const float* __restrict__ enc,       // (S,B,2E) f32
                                   const __bf16* __restrict__ attnW_bf, // (512,1536) bf16
                                   const float* __restrict__ hid_proj,  // (B,512)
                                   const float* __restrict__ v_w,       // (512)
                                   float* __restrict__ scores) {        // (B,S)
    __shared__ __align__(32) __bf16 As[16][E2 + 16]; // row 2080 B (32B mult)
    __shared__ float red[4][16];
    int b = blockIdx.y, stile = blockIdx.x;
    int tid = threadIdx.x;
    {   // stage 16 s-rows x 1024 e, float4 loads + packed cvt
        int row = tid >> 3;                  // 0..15
        int c0  = (tid & 7) * 128;           // 8 threads/row
        const float* src = enc + (size_t)(stile * 16 + row) * (BB * E2)
                               + (size_t)b * E2 + c0;
#pragma unroll
        for (int j = 0; j < 128; j += 4)
            *(v4bf*)&As[row][c0 + j] = cvt4(*(const v4f*)(src + j));
    }
    __syncthreads();

    int wave = tid >> 5, lane = tid & 31;
    int l15 = lane & 15;
    int kbA = (lane >> 4) * 16;
    int kbB = (lane >> 4) * 16;
    float part[8];
#pragma unroll
    for (int r = 0; r < 8; ++r) part[r] = 0.0f;

    for (int dt = wave; dt < 32; dt += 4) {
        const __bf16* wrow = attnW_bf + (size_t)(dt * 16 + l15) * FATT + DD;
        v8f acc = {};
#pragma unroll 2
        for (int k0 = 0; k0 < E2; k0 += 32) {
            v16bf af = load_a_frag(&As[l15][k0 + kbA]);
            v16bf bf = *(const v16bf*)(wrow + k0 + kbB);
            acc = wmma_bf16(af, bf, acc);
        }
        float hv = hid_proj[(size_t)b * DD + dt * 16 + l15];
        float vw = v_w[dt * 16 + l15];
#pragma unroll
        for (int r = 0; r < 8; ++r) part[r] += vw * tanhf(acc[r] + hv);
    }
#pragma unroll
    for (int r = 0; r < 8; ++r) {
        float v = part[r];
        v += __shfl_xor(v, 1);
        v += __shfl_xor(v, 2);
        v += __shfl_xor(v, 4);
        v += __shfl_xor(v, 8);
        part[r] = v;
    }
    if ((lane & 15) == 0) {
        int rowoff = (lane >> 4) * 8;
#pragma unroll
        for (int r = 0; r < 8; ++r) red[wave][rowoff + r] = part[r];
    }
    __syncthreads();
    if (tid < 16) {
        float s = red[0][tid] + red[1][tid] + red[2][tid] + red[3][tid];
        scores[(size_t)b * SS + stile * 16 + tid] = s;
    }
}

// ---------------------------------------------------------------------------
// K3: masked softmax over S. grid B, block 512 (16 waves).
// ---------------------------------------------------------------------------
__global__ void softmax_kernel(const float* __restrict__ scores,
                               const unsigned char* __restrict__ mask,
                               float* __restrict__ a_out) {
    __shared__ float red[16];
    int b = blockIdx.x, tid = threadIdx.x;
    int wave = tid >> 5, lane = tid & 31;
    float v = scores[(size_t)b * SS + tid];
    if (!mask[(size_t)b * SS + tid]) v = -INFINITY;
    float m = v;
#pragma unroll
    for (int off = 16; off; off >>= 1) m = fmaxf(m, __shfl_xor(m, off));
    if (lane == 0) red[wave] = m;
    __syncthreads();
    float mm = red[0];
#pragma unroll
    for (int i = 1; i < 16; ++i) mm = fmaxf(mm, red[i]);
    __syncthreads();
    float e = __expf(v - mm);
    float s = e;
#pragma unroll
    for (int off = 16; off; off >>= 1) s += __shfl_xor(s, off);
    if (lane == 0) red[wave] = s;
    __syncthreads();
    float ss = 0.0f;
#pragma unroll
    for (int i = 0; i < 16; ++i) ss += red[i];
    a_out[(size_t)b * SS + tid] = e / ss;
}

// ---------------------------------------------------------------------------
// K4: weighted[b,e] = sum_s a[b,s]*enc[s,b,e]. Bandwidth-bound VALU kernel.
// ---------------------------------------------------------------------------
__global__ void weighted_kernel(const float* __restrict__ enc,
                                const float* __restrict__ a,
                                __bf16* __restrict__ rnn_bf,
                                __bf16* __restrict__ zcat_bf) {
    int b = blockIdx.y;
    int e = (blockIdx.x * 256 + threadIdx.x) * 2;          // 0..1022
    const float* ab = a + (size_t)b * SS;
    const float* ep = enc + (size_t)b * E2 + e;
    float a0 = 0.0f, a1 = 0.0f;
#pragma unroll 4
    for (int s = 0; s < SS; ++s) {
        v2f x = *(const v2f*)(ep + (size_t)s * (BB * E2));
        float w = ab[s];
        a0 += w * x[0];
        a1 += w * x[1];
    }
    v2f r; r[0] = a0; r[1] = a1;
    v2bf y = cvt2(r);
    *(v2bf*)(rnn_bf + (size_t)b * FRNN + MM + e) = y;
    *(v2bf*)(zcat_bf + (size_t)b * FFC + DD + e) = y;
}

// ---------------------------------------------------------------------------
// K5c: GRU gates -> h_new (f32 output) + zcat_bf[:,0:512].
// ---------------------------------------------------------------------------
__global__ void gru_kernel(const float* __restrict__ gi,
                           const float* __restrict__ gh,
                           const float* __restrict__ dec,
                           __bf16* __restrict__ zcat_bf,
                           float* __restrict__ hnew) {
    int b = blockIdx.y;
    int d = (blockIdx.x * 256 + threadIdx.x) * 2;          // 0..510
    const float* gib = gi + (size_t)b * G3;
    const float* ghb = gh + (size_t)b * G3;
    v2f ir = *(const v2f*)(gib + d);
    v2f iz = *(const v2f*)(gib + DD + d);
    v2f in_ = *(const v2f*)(gib + 2 * DD + d);
    v2f hr = *(const v2f*)(ghb + d);
    v2f hz = *(const v2f*)(ghb + DD + d);
    v2f hn = *(const v2f*)(ghb + 2 * DD + d);
    v2f dc = *(const v2f*)(dec + (size_t)b * DD + d);
    v2f h;
#pragma unroll
    for (int i = 0; i < 2; ++i) {
        float r = 1.0f / (1.0f + __expf(-(ir[i] + hr[i])));
        float z = 1.0f / (1.0f + __expf(-(iz[i] + hz[i])));
        float n = tanhf(in_[i] + r * hn[i]);
        h[i] = (1.0f - z) * n + z * dc[i];
    }
    *(v2f*)(hnew + (size_t)b * DD + d) = h;
    *(v2bf*)(zcat_bf + (size_t)b * FFC + d) = cvt2(h);
}

// ---------------------------------------------------------------------------
// K6: prediction = zcat(128x2048,bf16) @ fc_W.T(50000x2048,f32) + fc_b.
// fc_W (410 MB) streamed from HBM exactly once. grid 3125 N-tiles of 16;
// block 128 = 4 waves; wave w owns rows 32w..32w+31 (2 M-tiles) so every
// B-fragment feeds 2 WMMAs.
// Async path: GLOBAL_LOAD_ASYNC_TO_LDS_B128 double-buffers the fp32 K-chunks
// straight into LDS (no VGPR round trip), overlapped with WMMA on the other
// buffer; ASYNCcnt-gated (wait <=16 keeps next chunk in flight).
// ---------------------------------------------------------------------------
__global__ void fc_kernel(const __bf16* __restrict__ zcat_bf,
                          const float* __restrict__ fcW,
                          const float* __restrict__ fcb,
                          float* __restrict__ pred) {
    int ntile = blockIdx.x;
    int tid = threadIdx.x, wave = tid >> 5, lane = tid & 31;
    int l15 = lane & 15;
    int kbA = (lane >> 4) * 16;
    int kbB = (lane >> 4) * 16;
    const __bf16* arow0 = zcat_bf + (size_t)(wave * 32 + l15) * FFC;
    const __bf16* arow1 = arow0 + (size_t)16 * FFC;
    v8f acc0 = {}, acc1 = {};
#if HAVE_ASYNC_LDS
    __shared__ __align__(32) float Fs[2][16][512 + 8];  // row 2080 B (32B mult)
    int srow = tid >> 3;                     // 0..15, 8 threads/row
    int sc0  = (tid & 7) * 64;
    const float* sbase = fcW + (size_t)(ntile * 16 + srow) * FFC + sc0;
    // prologue: async-stage chunk 0
#pragma unroll
    for (int j = 0; j < 64; j += 4)
        __builtin_amdgcn_global_load_async_to_lds_b128(
            (AS1 v4i*)(sbase + j), (AS3 v4i*)&Fs[0][srow][sc0 + j], 0, 0);
#pragma unroll
    for (int chunk = 0; chunk < 4; ++chunk) {
        if (chunk < 3) {   // keep next chunk in flight
            const float* nb = sbase + (chunk + 1) * 512;
#pragma unroll
            for (int j = 0; j < 64; j += 4)
                __builtin_amdgcn_global_load_async_to_lds_b128(
                    (AS1 v4i*)(nb + j),
                    (AS3 v4i*)&Fs[(chunk + 1) & 1][srow][sc0 + j], 0, 0);
            asm volatile("s_wait_asynccnt 0x10" ::: "memory");  // chunk done, next pending
        } else {
            asm volatile("s_wait_asynccnt 0x0" ::: "memory");
        }
        __syncthreads();
        const float* brow = &Fs[chunk & 1][l15][0];
        int kc = chunk * 512;
#pragma unroll 4
        for (int k0 = 0; k0 < 512; k0 += 32) {
            v16bf a0 = load_a_frag(arow0 + kc + k0 + kbA);
            v16bf a1 = load_a_frag(arow1 + kc + k0 + kbA);
            v16bf bf = load_b_frag_f32(brow + k0 + kbB);
            acc0 = wmma_bf16(a0, bf, acc0);
            acc1 = wmma_bf16(a1, bf, acc1);
        }
        __syncthreads();
    }
#else
    __shared__ __align__(32) __bf16 Bs[16][512 + 16];   // row 1056 B (32B mult)
    for (int kc = 0; kc < FFC; kc += 512) {
        {   // sync stage fc_W chunk -> bf16 LDS (float4 + packed cvt)
            int row = tid >> 3;
            int c0  = (tid & 7) * 64;
            const float* src = fcW + (size_t)(ntile * 16 + row) * FFC + kc + c0;
            __builtin_prefetch(src + 512, 0, 0);
#pragma unroll
            for (int j = 0; j < 64; j += 4)
                *(v4bf*)&Bs[row][c0 + j] = cvt4(*(const v4f*)(src + j));
        }
        __syncthreads();
#pragma unroll 4
        for (int k0 = 0; k0 < 512; k0 += 32) {
            v16bf a0 = load_a_frag(arow0 + kc + k0 + kbA);
            v16bf a1 = load_a_frag(arow1 + kc + k0 + kbA);
            v16bf bf = *(const v16bf*)&Bs[l15][k0 + kbB];
            acc0 = wmma_bf16(a0, bf, acc0);
            acc1 = wmma_bf16(a1, bf, acc1);
        }
        __syncthreads();
    }
#endif
    int rbase = (lane >> 4) * 8;
    float bv = fcb[ntile * 16 + l15];
#pragma unroll
    for (int r = 0; r < 8; ++r) {
        int row0 = wave * 32 + rbase + r;
        pred[(size_t)row0 * VV + ntile * 16 + l15] = acc0[r] + bv;
        pred[(size_t)(row0 + 16) * VV + ntile * 16 + l15] = acc1[r] + bv;
    }
}

// ---------------------------------------------------------------------------
extern "C" void kernel_launch(void* const* d_in, const int* in_sizes, int n_in,
                              void* d_out, int out_size, void* d_ws, size_t ws_size,
                              hipStream_t stream) {
    const int*           input     = (const int*)d_in[0];
    const float*         dec_state = (const float*)d_in[1];
    const float*         enc       = (const float*)d_in[2];   // (S,B,2E)
    const unsigned char* mask      = (const unsigned char*)d_in[3];
    const float*         emb       = (const float*)d_in[4];
    const float*         attn_W    = (const float*)d_in[5];   // (512,1536)
    const float*         attn_b    = (const float*)d_in[6];
    const float*         v_w       = (const float*)d_in[7];
    const float*         gru_Wih   = (const float*)d_in[8];   // (1536,1536)
    const float*         gru_bih   = (const float*)d_in[9];
    const float*         gru_Whh   = (const float*)d_in[10];  // (1536,512)
    const float*         gru_bhh   = (const float*)d_in[11];
    const float*         fcW       = (const float*)d_in[12];  // (50000,2048)
    const float*         fcb       = (const float*)d_in[13];

    float* out  = (float*)d_out;
    float* pred = out;                        // B*V  = 6,400,000
    float* hnew = out + (size_t)BB * VV;      // B*D  = 65,536
    float* a    = hnew + (size_t)BB * DD;     // B*S  = 65,536

    // workspace layout (byte offsets, all 256B-aligned)
    char* w = (char*)d_ws;
    float*  hid_proj = (float*)(w + 0);              //  262,144 B
    float*  scores   = (float*)(w + 262144);         //  262,144 B
    float*  gi       = (float*)(w + 524288);         //  786,432 B
    float*  gh       = (float*)(w + 1310720);        //  786,432 B
    __bf16* dec_bf   = (__bf16*)(w + 2097152);       //  131,072 B
    __bf16* rnn_bf   = (__bf16*)(w + 2228224);       //  393,216 B
    __bf16* zcat_bf  = (__bf16*)(w + 2621440);       //  524,288 B
    __bf16* attnW_bf = (__bf16*)(w + 3145728);       // 1,572,864 B
    __bf16* wih_bf   = (__bf16*)(w + 4718592);       // 4,718,592 B
    __bf16* whh_bf   = (__bf16*)(w + 9437184);       // 1,572,864 B  (~10.5 MB total)

    // Pre-convert reused GEMM operands to bf16 (packed hardware cvt)
    cvt_kernel<<<(DD * FATT / 4 + 255) / 256, 256, 0, stream>>>(attn_W, attnW_bf, DD * FATT / 4);
    cvt_kernel<<<(G3 * FRNN / 4 + 255) / 256, 256, 0, stream>>>(gru_Wih, wih_bf, G3 * FRNN / 4);
    cvt_kernel<<<(G3 * DD / 4 + 255) / 256, 256, 0, stream>>>(gru_Whh, whh_bf, G3 * DD / 4);
    cvt_kernel<<<(BB * DD / 4 + 255) / 256, 256, 0, stream>>>(dec_state, dec_bf, BB * DD / 4);

    // K0: embedding gather
    embed_kernel<<<dim3(1, BB), 256, 0, stream>>>(input, emb, rnn_bf, zcat_bf);
    // K1: hid_proj = dec_state @ attn_W[:, :512].T + attn_b
    gemm_bf16<<<dim3(DD / 16, BB / 16), 32, 0, stream>>>(
        dec_bf, DD, attnW_bf, FATT, attn_b, hid_proj, DD, DD);
    // K2: fused energy/tanh/v./reduce -> scores
    attn_scores_kernel<<<dim3(SS / 16, BB), 128, 0, stream>>>(
        enc, attnW_bf, hid_proj, v_w, scores);
    // K3: masked softmax -> a (output)
    softmax_kernel<<<BB, SS, 0, stream>>>(scores, mask, a);
    // K4: weighted = a @ enc
    weighted_kernel<<<dim3(2, BB), 256, 0, stream>>>(enc, a, rnn_bf, zcat_bf);
    // K5a: gi = rnn_in @ gru_Wih.T + bih
    gemm_bf16<<<dim3(G3 / 16, BB / 16), 32, 0, stream>>>(
        rnn_bf, FRNN, wih_bf, FRNN, gru_bih, gi, G3, FRNN);
    // K5b: gh = dec_state @ gru_Whh.T + bhh
    gemm_bf16<<<dim3(G3 / 16, BB / 16), 32, 0, stream>>>(
        dec_bf, DD, whh_bf, DD, gru_bhh, gh, G3, DD);
    // K5c: GRU gates -> h_new (output) + zcat_bf[:,0:512]
    gru_kernel<<<dim3(1, BB), 256, 0, stream>>>(gi, gh, dec_state, zcat_bf, hnew);
    // K6: prediction = zcat @ fc_W.T + fc_b
    fc_kernel<<<VV / 16, 128, 0, stream>>>(zcat_bf, fcW, fcb, pred);
}
